// CSLRTransformer_77060303225154
// MI455X (gfx1250) — matrix operand order repaired and checked by reading, not compiled, not run on hardware
//
#include <hip/hip_runtime.h>
#include <hip/hip_bf16.h>

// ---------------------------------------------------------------------------
// Model dims
// ---------------------------------------------------------------------------
#define BB   4
#define TT   1024
#define INDIM 231
#define DD   512
#define HH   8
#define DH   64
#define NLY  8
#define DFF  2048
#define T2   512          // after AvgPool1d(2)
#define NC   1296
#define EPSV 1e-5f

typedef __attribute__((ext_vector_type(16))) _Float16 v16h;
typedef __attribute__((ext_vector_type(2)))  __fp16   v2hp;   // cvt_pkrtz result type
typedef __attribute__((ext_vector_type(8)))  float    v8f;

__device__ __forceinline__ float gelu_exact(float v) {
    return 0.5f * v * (1.f + erff(v * 0.70710678118654752f));
}

// ---------------------------------------------------------------------------
// Generic WMMA GEMM:  Out[m,n] = fuse( alpha * sum_k A[m,k]*B(k,n) + bias[n] ) (+ res)
//   TRANSB==0 : B given as W[N,K] row-major (ldb strides n)  -> x @ W^T
//   TRANSB==1 : B given as B[K,N] row-major (ldb strides k)  -> x @ B
// Batched via blockIdx.z with (bb,hh) = (z/nH, z%nH) offset decomposition.
// Block: 256 thr = 8 wave32, tile 128(M) x 64(N), K-step 32, LDS double-buffered.
// Wave grid 4(M) x 2(N); each wave issues 2x2 v_wmma_f32_16x16x32_f16 per K-step.
// Fast path: unguarded float4 loads + v_cvt_pk_rtz packing; slow path is
// branch-free (clamped address + select) for ragged edges (embed K=231, fc2 N=1296).
// ---------------------------------------------------------------------------
#define BM 128
#define BN 64
#define BK 32

template<int BIAS, int ACT, int RES, int TRANSB>
__global__ __launch_bounds__(256)
void gemm_wmma(const float* __restrict__ A,  long lda,
               const float* __restrict__ Bm, long ldb,
               const float* __restrict__ bias,
               const float* __restrict__ res, long ldr,
               float*       __restrict__ Out, long ldo,
               int M, int N, int K, float alpha, int nH,
               long aB, long aH, long bB, long bH,
               long oB, long oH, long rB, long rH)
{
    const int z  = blockIdx.z;
    const int zb = z / nH, zh = z % nH;
    A   += (long)zb * aB + (long)zh * aH;
    Bm  += (long)zb * bB + (long)zh * bH;
    Out += (long)zb * oB + (long)zh * oH;
    if (RES) res += (long)zb * rB + (long)zh * rH;

    __shared__ _Float16 As[2][BM][BK + 8];
    __shared__ _Float16 Bs[2][BN][BK + 8];

    const int tid  = threadIdx.x;
    const int wave = tid >> 5;
    const int lane = tid & 31;
    const int wm   = wave & 3;      // wave M index 0..3
    const int wn   = wave >> 2;     // wave N index 0..1
    const int grp  = lane >> 4;     // half-wave group
    const int lrow = lane & 15;

    const int m0 = blockIdx.x * BM;
    const int n0 = blockIdx.y * BN;

    // uniform (scalar) alignment/in-bounds classification for the fast path
    const bool vecOK = ((lda & 3) == 0) && ((ldb & 3) == 0) &&
                       (m0 + BM <= M) && (n0 + BN <= N);

    const int ksteps = (K + BK - 1) / BK;

    // ---- tile staging (double buffered) ----
    auto stage = [&](int ks, int buf) {
        const int k0 = ks * BK;
        if (vecOK && (k0 + BK) <= K) {
            // -------- fast path: float4 loads, packed f16 LDS stores --------
            if (k0 + BK < K)   // prefetch next K-tile of A (global_prefetch_b8)
                __builtin_prefetch(A + (long)(m0 + (tid >> 3)) * lda + k0 + BK, 0, 1);
            #pragma unroll
            for (int i = 0; i < 4; ++i) {                 // A: 128x32 = 1024 float4
                int idx = tid + i * 256;
                int r = idx >> 3;                          // 8 float4 per row
                int c = (idx & 7) << 2;
                const float4 v = *(const float4*)(A + (long)(m0 + r) * lda + k0 + c);
                *(v2hp*)&As[buf][r][c]     = __builtin_amdgcn_cvt_pkrtz(v.x, v.y);
                *(v2hp*)&As[buf][r][c + 2] = __builtin_amdgcn_cvt_pkrtz(v.z, v.w);
            }
            if (TRANSB == 0) {
                #pragma unroll
                for (int i = 0; i < 2; ++i) {             // B: 64x32 = 512 float4
                    int idx = tid + i * 256;
                    int r = idx >> 3;
                    int c = (idx & 7) << 2;
                    const float4 v = *(const float4*)(Bm + (long)(n0 + r) * ldb + k0 + c);
                    *(v2hp*)&Bs[buf][r][c]     = __builtin_amdgcn_cvt_pkrtz(v.x, v.y);
                    *(v2hp*)&Bs[buf][r][c + 2] = __builtin_amdgcn_cvt_pkrtz(v.z, v.w);
                }
            } else {
                #pragma unroll
                for (int i = 0; i < 2; ++i) {             // B[K,N]: coalesce along n
                    int idx = tid + i * 256;
                    int kk = idx >> 4;                     // 16 float4 per k-row
                    int nn = (idx & 15) << 2;
                    const float4 v = *(const float4*)(Bm + (long)(k0 + kk) * ldb + n0 + nn);
                    Bs[buf][nn    ][kk] = (_Float16)v.x;
                    Bs[buf][nn + 1][kk] = (_Float16)v.y;
                    Bs[buf][nn + 2][kk] = (_Float16)v.z;
                    Bs[buf][nn + 3][kk] = (_Float16)v.w;
                }
            }
        } else {
            // -------- branch-free guarded path (tile edges only) --------
            #pragma unroll
            for (int i = 0; i < 16; ++i) {
                int idx = tid + i * 256;
                int r = idx >> 5, c = idx & 31;
                int gm = m0 + r, gk = k0 + c;
                int cm = gm < M ? gm : M - 1;
                int ck = gk < K ? gk : K - 1;
                float v = A[(long)cm * lda + ck];
                v = (gm < M && gk < K) ? v : 0.f;
                As[buf][r][c] = (_Float16)v;
            }
            #pragma unroll
            for (int i = 0; i < 8; ++i) {
                int idx = tid + i * 256;
                int r, c, gn, gk;
                if (TRANSB == 0) { r = idx >> 5; c = idx & 31; }
                else             { c = idx >> 6; r = idx & 63; }
                gn = n0 + r; gk = k0 + c;
                int cn = gn < N ? gn : N - 1;
                int ck = gk < K ? gk : K - 1;
                float v = (TRANSB == 0) ? Bm[(long)cn * ldb + ck]
                                        : Bm[(long)ck * ldb + cn];
                v = (gn < N && gk < K) ? v : 0.f;
                Bs[buf][r][c] = (_Float16)v;
            }
        }
    };

    v8f acc[2][2] = {};

    int buf = 0;
    stage(0, 0);
    for (int ks = 0; ks < ksteps; ++ks) {
        __syncthreads();
        if (ks + 1 < ksteps) stage(ks + 1, buf ^ 1);   // overlap with compute

        #pragma unroll
        for (int ti = 0; ti < 2; ++ti) {
            v16h a;
            const int ar = wm * 32 + ti * 16 + lrow;       // A: lane holds row M
            #pragma unroll
            for (int j = 0; j < 8; ++j) {                  // VGPR j K-pairs
                int kk = ((j & 4) << 2) + ((j & 3) << 1) + (grp << 3);
                a[2 * j]     = As[buf][ar][kk];
                a[2 * j + 1] = As[buf][ar][kk + 1];
            }
            #pragma unroll
            for (int tj = 0; tj < 2; ++tj) {
                v16h bf;
                const int br = wn * 32 + tj * 16 + lrow;   // B: lane holds col N
                #pragma unroll
                for (int j = 0; j < 8; ++j) {
                    int kk = ((j & 4) << 2) + ((j & 3) << 1) + (grp << 3);
                    bf[2 * j]     = Bs[buf][br][kk];
                    bf[2 * j + 1] = Bs[buf][br][kk + 1];
                }
                acc[ti][tj] = __builtin_amdgcn_wmma_f32_16x16x32_f16(
                    false, a, false, bf, (short)0, acc[ti][tj], false, false);
            }
        }
        buf ^= 1;
    }

    // ---- epilogue: scale, bias, activation, residual, store ----
    #pragma unroll
    for (int ti = 0; ti < 2; ++ti)
        #pragma unroll
        for (int tj = 0; tj < 2; ++tj)
            #pragma unroll
            for (int r = 0; r < 8; ++r) {
                int gm = m0 + wm * 32 + ti * 16 + r + (grp << 3);
                int gn = n0 + wn * 32 + tj * 16 + lrow;
                if (gm < M && gn < N) {
                    float v = acc[ti][tj][r] * alpha;
                    if (BIAS) v += bias[gn];
                    if (ACT == 1) v = v > 0.f ? v : 0.f;
                    if (ACT == 2) v = gelu_exact(v);
                    if (RES) v += res[(long)gm * ldr + gn];
                    Out[(long)gm * ldo + gn] = v;
                }
            }
}

// ---------------------------------------------------------------------------
// LayerNorm, wave-per-row (D=512 -> 16 elems/lane in wave32)
// ---------------------------------------------------------------------------
__global__ __launch_bounds__(256)
void ln_rows(const float* __restrict__ x, const float* __restrict__ g,
             const float* __restrict__ b, float* __restrict__ y, int Dc)
{
    int row  = blockIdx.x * (blockDim.x >> 5) + (threadIdx.x >> 5);
    int lane = threadIdx.x & 31;
    const float* xr = x + (long)row * Dc;
    float s = 0.f, s2 = 0.f;
    for (int i = lane; i < Dc; i += 32) { float v = xr[i]; s += v; s2 += v * v; }
    #pragma unroll
    for (int o = 16; o; o >>= 1) { s += __shfl_xor(s, o, 32); s2 += __shfl_xor(s2, o, 32); }
    float mean = s / Dc;
    float var  = s2 / Dc - mean * mean;
    float inv  = rsqrtf(var + EPSV);
    float* yr = y + (long)row * Dc;
    for (int i = lane; i < Dc; i += 32)
        yr[i] = (xr[i] - mean) * inv * g[i] + b[i];
}

// LN + sinusoidal positional encoding (row = b*T + t)
__global__ __launch_bounds__(256)
void ln_pe_rows(const float* __restrict__ x, const float* __restrict__ g,
                const float* __restrict__ b, float* __restrict__ y)
{
    int row  = blockIdx.x * (blockDim.x >> 5) + (threadIdx.x >> 5);
    int lane = threadIdx.x & 31;
    int t    = row % TT;
    const float* xr = x + (long)row * DD;
    float s = 0.f, s2 = 0.f;
    for (int i = lane; i < DD; i += 32) { float v = xr[i]; s += v; s2 += v * v; }
    #pragma unroll
    for (int o = 16; o; o >>= 1) { s += __shfl_xor(s, o, 32); s2 += __shfl_xor(s2, o, 32); }
    float mean = s / DD;
    float inv  = rsqrtf(s2 / DD - mean * mean + EPSV);
    float* yr = y + (long)row * DD;
    for (int i = lane; i < DD; i += 32) {
        float ln = (xr[i] - mean) * inv * g[i] + b[i];
        float fr = __powf(10000.f, -(float)(i & ~1) * (1.f / (float)DD));
        float ar = (float)t * fr;
        float pe = (i & 1) ? __cosf(ar) : __sinf(ar);
        yr[i] = ln + pe;
    }
}

// ---------------------------------------------------------------------------
// Softmax, wave-per-row over 1024 columns (3 strided passes)
// ---------------------------------------------------------------------------
__global__ __launch_bounds__(256)
void softmax_rows(float* __restrict__ s, int Ncols)
{
    int row  = blockIdx.x * (blockDim.x >> 5) + (threadIdx.x >> 5);
    int lane = threadIdx.x & 31;
    float* r = s + (long)row * Ncols;
    float mx = -3.4e38f;
    for (int i = lane; i < Ncols; i += 32) mx = fmaxf(mx, r[i]);
    #pragma unroll
    for (int o = 16; o; o >>= 1) mx = fmaxf(mx, __shfl_xor(mx, o, 32));
    float sum = 0.f;
    for (int i = lane; i < Ncols; i += 32) { float e = __expf(r[i] - mx); r[i] = e; sum += e; }
    #pragma unroll
    for (int o = 16; o; o >>= 1) sum += __shfl_xor(sum, o, 32);
    float inv = 1.f / sum;
    for (int i = lane; i < Ncols; i += 32) r[i] *= inv;
}

// ---------------------------------------------------------------------------
// Elementwise / layout kernels
// ---------------------------------------------------------------------------
__global__ void add_inplace(float* __restrict__ x, const float* __restrict__ s, long n)
{
    long i = (long)blockIdx.x * blockDim.x + threadIdx.x;
    if (i < n) x[i] += s[i];
}

// x [B,T,D] -> pooled [B,D,T/2] (AvgPool1d(2,2) fused with NTD->NDT transpose)
__global__ void pool_transpose(const float* __restrict__ x, float* __restrict__ o)
{
    long idx = (long)blockIdx.x * blockDim.x + threadIdx.x;
    if (idx >= (long)BB * DD * T2) return;
    int t = idx % T2;
    int d = (idx / T2) % DD;
    int b = idx / ((long)T2 * DD);
    const float* xb = x + (long)b * TT * DD;
    o[idx] = 0.5f * (xb[(long)(2 * t) * DD + d] + xb[(long)(2 * t + 1) * DD + d]);
}

// conv1d (NCH) + eval BatchNorm + exact GELU; one thread per (b,co,t)
__global__ void conv_bn_gelu(const float* __restrict__ in, const float* __restrict__ w,
                             const float* __restrict__ bias,
                             const float* __restrict__ bng, const float* __restrict__ bnb,
                             const float* __restrict__ bnm, const float* __restrict__ bnv,
                             float* __restrict__ out, int Kw, int pad)
{
    long idx = (long)blockIdx.x * blockDim.x + threadIdx.x;
    if (idx >= (long)BB * DD * T2) return;
    int t  = idx % T2;
    int co = (idx / T2) % DD;
    int b  = idx / ((long)T2 * DD);
    const float* ib = in + (long)b * DD * T2;
    const float* wc = w + (long)co * DD * Kw;
    float acc = bias[co];
    for (int ci = 0; ci < DD; ++ci) {
        const float* row = ib + (long)ci * T2;
        const float* wr  = wc + ci * Kw;
        for (int j = 0; j < Kw; ++j) {
            int tt = t + j - pad;
            if (tt >= 0 && tt < T2) acc += row[tt] * wr[j];
        }
    }
    float v = (acc - bnm[co]) * rsqrtf(bnv[co] + EPSV) * bng[co] + bnb[co];
    out[idx] = gelu_exact(v);
}

// in [B,D,T2] -> out [B*T2, D]
__global__ void transpose_dt(const float* __restrict__ in, float* __restrict__ out)
{
    long idx = (long)blockIdx.x * blockDim.x + threadIdx.x;
    if (idx >= (long)BB * T2 * DD) return;
    int d = idx % DD;
    int t = (idx / DD) % T2;
    int b = idx / ((long)DD * T2);
    out[idx] = in[(long)b * DD * T2 + (long)d * T2 + t];
}

// ---------------------------------------------------------------------------
// Host launcher
// ---------------------------------------------------------------------------
static inline dim3 ggrid(int M, int N, int Z) {
    return dim3((M + BM - 1) / BM, (N + BN - 1) / BN, Z);
}

extern "C" void kernel_launch(void* const* d_in, const int* in_sizes, int n_in,
                              void* d_out, int out_size, void* d_ws, size_t ws_size,
                              hipStream_t stream)
{
    const float* poses   = (const float*)d_in[0];
    const float* embed_w = (const float*)d_in[1];
    const float* embed_b = (const float*)d_in[2];
    const float* ln0_g   = (const float*)d_in[3];
    const float* ln0_b   = (const float*)d_in[4];
    const float* inw     = (const float*)d_in[5];
    const float* inb     = (const float*)d_in[6];
    const float* outw    = (const float*)d_in[7];
    const float* outb    = (const float*)d_in[8];
    const float* ln1g    = (const float*)d_in[9];
    const float* ln1b    = (const float*)d_in[10];
    const float* ln2g    = (const float*)d_in[11];
    const float* ln2b    = (const float*)d_in[12];
    const float* ff1w    = (const float*)d_in[13];
    const float* ff1b    = (const float*)d_in[14];
    const float* ff2w    = (const float*)d_in[15];
    const float* ff2b    = (const float*)d_in[16];
    const float* conv1w  = (const float*)d_in[17];
    const float* conv1b  = (const float*)d_in[18];
    const float* bn1g    = (const float*)d_in[19];
    const float* bn1b    = (const float*)d_in[20];
    const float* bn1m    = (const float*)d_in[21];
    const float* bn1v    = (const float*)d_in[22];
    const float* conv2w  = (const float*)d_in[23];
    const float* conv2b  = (const float*)d_in[24];
    const float* bn2g    = (const float*)d_in[25];
    const float* bn2b    = (const float*)d_in[26];
    const float* bn2m    = (const float*)d_in[27];
    const float* bn2v    = (const float*)d_in[28];
    const float* fc1w    = (const float*)d_in[29];
    const float* fc1b    = (const float*)d_in[30];
    const float* fc2w    = (const float*)d_in[31];
    const float* fc2b    = (const float*)d_in[32];

    const long MTOK = (long)BB * TT;          // 4096 token rows

    // workspace carve-up (f32)
    float* p      = (float*)d_ws;
    float* x      = p;  p += MTOK * DD;           // activations [4096,512]
    float* hbuf   = p;  p += MTOK * DD;           // LN output / embed tmp
    float* qkv    = p;  p += MTOK * 3 * DD;       // [4096,1536]
    float* ctx    = p;  p += MTOK * DD;
    float* ffbuf  = p;  p += MTOK * DFF;          // [4096,2048]
    float* save   = p;  p += MTOK * DD;           // block residual save
    float* scores = p;  p += (long)BB * HH * TT * TT;   // [32,1024,1024]
    float* pooled = p;  p += (long)BB * DD * T2;
    float* cbuf   = p;  p += (long)BB * DD * T2;
    float* fc1o   = p;  p += (long)BB * T2 * (DD / 2);

    const long xBytes = MTOK * DD * sizeof(float);

    // ---- embed: poses[4096,231] @ embed_w[512,231]^T + b -> hbuf ----
    gemm_wmma<1,0,0,0><<<ggrid(MTOK, DD, 1), 256, 0, stream>>>(
        poses, INDIM, embed_w, INDIM, embed_b, nullptr, 0, hbuf, DD,
        MTOK, DD, INDIM, 1.f, 1, 0,0,0,0,0,0,0,0);
    // ---- LN0 + positional encoding -> x ----
    ln_pe_rows<<<MTOK / 8, 256, 0, stream>>>(hbuf, ln0_g, ln0_b, x);

    const long qkvRow = 3 * DD;               // 1536
    for (int l = 0; l < NLY; ++l) {
        if (l == 2 || l == 4 || l == 6)
            (void)hipMemcpyAsync(save, x, xBytes, hipMemcpyDeviceToDevice, stream);

        // h = LN1(x)
        ln_rows<<<MTOK / 8, 256, 0, stream>>>(x, ln1g + l * DD, ln1b + l * DD, hbuf, DD);

        // qkv = h @ inw[l]^T + inb[l]
        gemm_wmma<1,0,0,0><<<ggrid(MTOK, 3 * DD, 1), 256, 0, stream>>>(
            hbuf, DD, inw + (long)l * 3 * DD * DD, DD, inb + (long)l * 3 * DD,
            nullptr, 0, qkv, qkvRow, MTOK, 3 * DD, DD, 1.f, 1, 0,0,0,0,0,0,0,0);

        // scores[bh] = Q @ K^T / 8   (batched over 32 (b,h) via stride decomposition)
        gemm_wmma<0,0,0,0><<<ggrid(TT, TT, BB * HH), 256, 0, stream>>>(
            qkv, qkvRow,                   // Q slice
            qkv + DD, qkvRow,              // K slice (W[N,K] form)
            nullptr, nullptr, 0,
            scores, TT,
            TT, TT, DH, 0.125f, HH,
            (long)TT * qkvRow, DH,         // A: per-batch, per-head
            (long)TT * qkvRow, DH,         // B: per-batch, per-head
            (long)HH * TT * TT, (long)TT * TT,  // Out
            0, 0);

        softmax_rows<<<(BB * HH * TT) / 8, 256, 0, stream>>>(scores, TT);

        // ctx[bh] = A @ V   (B given as [K,N] row-major -> TRANSB=1)
        gemm_wmma<0,0,0,1><<<ggrid(TT, DH, BB * HH), 256, 0, stream>>>(
            scores, TT,
            qkv + 2 * DD, qkvRow,
            nullptr, nullptr, 0,
            ctx, DD,
            TT, DH, TT, 1.f, HH,
            (long)HH * TT * TT, (long)TT * TT,
            (long)TT * qkvRow, DH,
            (long)TT * DD, DH,
            0, 0);

        // x = x + ctx @ outw[l]^T + outb[l]
        gemm_wmma<1,0,1,0><<<ggrid(MTOK, DD, 1), 256, 0, stream>>>(
            ctx, DD, outw + (long)l * DD * DD, DD, outb + (long)l * DD,
            x, DD, x, DD, MTOK, DD, DD, 1.f, 1, 0,0,0,0,0,0,0,0);

        // h2 = LN2(x)
        ln_rows<<<MTOK / 8, 256, 0, stream>>>(x, ln2g + l * DD, ln2b + l * DD, hbuf, DD);

        // ff = relu(h2 @ ff1w[l]^T + ff1b[l])
        gemm_wmma<1,1,0,0><<<ggrid(MTOK, DFF, 1), 256, 0, stream>>>(
            hbuf, DD, ff1w + (long)l * DFF * DD, DD, ff1b + (long)l * DFF,
            nullptr, 0, ffbuf, DFF, MTOK, DFF, DD, 1.f, 1, 0,0,0,0,0,0,0,0);

        // x = x + ff @ ff2w[l]^T + ff2b[l]
        gemm_wmma<1,0,1,0><<<ggrid(MTOK, DD, 1), 256, 0, stream>>>(
            ffbuf, DFF, ff2w + (long)l * DD * DFF, DFF, ff2b + (long)l * DD,
            x, DD, x, DD, MTOK, DD, DFF, 1.f, 1, 0,0,0,0,0,0,0,0);

        if (l == 3 || l == 5 || l == 7)
            add_inplace<<<(MTOK * DD + 255) / 256, 256, 0, stream>>>(x, save, MTOK * DD);
    }

    // ---- head: pool+transpose, conv-bn-gelu x2, transpose, fc1(gelu), fc2 ----
    const long nchw = (long)BB * DD * T2;
    pool_transpose<<<(nchw + 255) / 256, 256, 0, stream>>>(x, pooled);
    conv_bn_gelu<<<(nchw + 255) / 256, 256, 0, stream>>>(
        pooled, conv1w, conv1b, bn1g, bn1b, bn1m, bn1v, cbuf, 5, 2);
    conv_bn_gelu<<<(nchw + 255) / 256, 256, 0, stream>>>(
        cbuf, conv2w, conv2b, bn2g, bn2b, bn2m, bn2v, pooled, 3, 1);
    transpose_dt<<<(nchw + 255) / 256, 256, 0, stream>>>(pooled, cbuf); // cbuf = [B*T2, D]

    const long MFC = (long)BB * T2;  // 2048
    gemm_wmma<1,2,0,0><<<ggrid(MFC, DD / 2, 1), 256, 0, stream>>>(
        cbuf, DD, fc1w, DD, fc1b, nullptr, 0, fc1o, DD / 2,
        MFC, DD / 2, DD, 1.f, 1, 0,0,0,0,0,0,0,0);
    gemm_wmma<1,0,0,0><<<ggrid(MFC, NC, 1), 256, 0, stream>>>(
        fc1o, DD / 2, fc2w, DD / 2, fc2b, nullptr, 0, (float*)d_out, NC,
        MFC, NC, DD / 2, 1.f, 1, 0,0,0,0,0,0,0,0);
}
